// AttentionFFNBlock_4569845202962
// MI455X (gfx1250) — compile-verified, matrix operand order checked by reading, not run on hardware
//
#include <hip/hip_runtime.h>

// ---------------------------------------------------------------------------
// AttentionFFNBlock for MI455X (gfx1250, wave32, WMMA).
// All heavy math goes through V_WMMA_F32_16X16X32_BF16 (bf16 in, f32 accum).
// GEMM staging uses GLOBAL_LOAD_ASYNC_TO_LDS_B128 (ASYNCcnt) — CDNA5 async
// memory->LDS path — instead of a VGPR round trip.
// ---------------------------------------------------------------------------

typedef unsigned short u16;
typedef unsigned int   u32;
typedef __attribute__((ext_vector_type(16))) __bf16 v16bf;
typedef __attribute__((ext_vector_type(8)))  float  v8f;

struct U4x2 { uint4 a, b; };

static __device__ __forceinline__ u16 f2bf(float f) {
  u32 u = __float_as_uint(f);
  u32 r = u + 0x7FFFu + ((u >> 16) & 1u);   // round-to-nearest-even
  return (u16)(r >> 16);
}

static __device__ __forceinline__ v8f vzero() {
  v8f z = {0.f, 0.f, 0.f, 0.f, 0.f, 0.f, 0.f, 0.f};
  return z;
}

// Load a 16x32 bf16 A/B fragment (CDNA5 16-bit layout): lane holds row (l&15),
// K chunks at (l>>4)*8 and +16.  Caller passes p already offset by the K-chunk.
static __device__ __forceinline__ v16bf load_frag(const u16* p) {
  U4x2 r;
  r.a = *(const uint4*)(p);
  r.b = *(const uint4*)(p + 16);
  return __builtin_bit_cast(v16bf, r);
}

static __device__ __forceinline__ v8f wmma_bf16(v16bf a, v16bf b, v8f c) {
  return __builtin_amdgcn_wmma_f32_16x16x32_bf16(false, a, false, b, (short)0,
                                                 c, false, false);
}

// Async global->LDS 16B copy (per lane).  LDS offset = low 32 bits of the
// generic shared pointer (aperture maps LDS by truncation); global address is
// the 64-bit pointer in a VGPR pair.  Tracked by ASYNCcnt.
static __device__ __forceinline__ void async_ld_b128(u16* lds, const u16* g) {
  asm volatile("global_load_async_to_lds_b128 %0, %1, off"
               :
               : "v"((u32)(uintptr_t)lds), "v"(g)
               : "memory");
}
static __device__ __forceinline__ void wait_async0() {
  asm volatile("s_wait_asynccnt 0x0" ::: "memory");
}

static __device__ __forceinline__ float rmax16(float x) {
#pragma unroll
  for (int m = 1; m < 16; m <<= 1) x = fmaxf(x, __shfl_xor(x, m, 32));
  return x;
}
static __device__ __forceinline__ float rsum16(float x) {
#pragma unroll
  for (int m = 1; m < 16; m <<= 1) x += __shfl_xor(x, m, 32);
  return x;
}

// ---------------------------------------------------------------------------
// fp32 -> bf16 elementwise
// ---------------------------------------------------------------------------
__global__ void k_f32_to_bf16(const float* __restrict__ in,
                              u16* __restrict__ out, int n) {
  for (int i = blockIdx.x * blockDim.x + threadIdx.x; i < n;
       i += gridDim.x * blockDim.x)
    out[i] = f2bf(in[i]);
}

// ---------------------------------------------------------------------------
// Weight transpose + convert: Wt[n][k] = bf16(W[k][n]).  W is [K][N].
// block (32,8), grid (K/32, N/32)
// ---------------------------------------------------------------------------
__global__ __launch_bounds__(256) void k_wtrans(const float* __restrict__ W,
                                                u16* __restrict__ Wt,
                                                int K, int N) {
  __shared__ float tile[32][33];
  const int tx = threadIdx.x, ty = threadIdx.y;
  const int k0 = blockIdx.x * 32, n0 = blockIdx.y * 32;
#pragma unroll
  for (int j = 0; j < 4; ++j)
    tile[ty + j * 8][tx] = W[(size_t)(k0 + ty + j * 8) * N + n0 + tx];
  __syncthreads();
#pragma unroll
  for (int j = 0; j < 4; ++j)
    Wt[(size_t)(n0 + ty + j * 8) * K + k0 + tx] = f2bf(tile[tx][ty + j * 8]);
}

// ---------------------------------------------------------------------------
// V transpose (bf16): Vt[(b*H+h)*Hd + d][l] = V[b*L+l][h*Hd + d]
// block (32,8), grid (L/32, Hd/32, B*H)
// ---------------------------------------------------------------------------
__global__ __launch_bounds__(256) void k_vtrans(const u16* __restrict__ V,
                                                u16* __restrict__ Vt,
                                                int H, int L, int D, int Hd) {
  __shared__ u16 tile[32][33];
  const int tx = threadIdx.x, ty = threadIdx.y;
  const int l0 = blockIdx.x * 32, d0 = blockIdx.y * 32;
  const int bh = blockIdx.z;
  const int b = bh / H, h = bh % H;
#pragma unroll
  for (int j = 0; j < 4; ++j)
    tile[ty + j * 8][tx] =
        V[(size_t)(b * L + l0 + ty + j * 8) * D + h * Hd + d0 + tx];
  __syncthreads();
#pragma unroll
  for (int j = 0; j < 4; ++j)
    Vt[(size_t)(bh * Hd + d0 + ty + j * 8) * L + l0 + tx] =
        tile[tx][ty + j * 8];
}

// ---------------------------------------------------------------------------
// Tiled WMMA GEMM: C[M][N] = act(A[M][K] @ Bt[N][K]^T + bias[N])
// A, Bt bf16 row-major; Bt is the transposed weight so both operands use the
// same K-striped fragment load.
// 256 threads = 8 waves (4x2); 128x128x32 block tile; each wave owns a 32x64
// quadrant = 2x4 WMMA tiles -> 8 WMMAs per 12 ds_load_b128 per K-step.
// Staging is async (global_load_async_to_lds_b128 + s_wait_asynccnt).
// act: 0 = identity, 1 = exact GELU.  Writes Cb (bf16) and/or Cf (f32).
// ---------------------------------------------------------------------------
#define BM 128
#define BN 128
#define BK 32
#define TM 2
#define TN 4
#define LDT 40  // LDS row stride in halves (80B, 16B-aligned, bank-skewed)

__global__ __launch_bounds__(256) void k_gemm_bf16(
    const u16* __restrict__ A, const u16* __restrict__ Bt,
    const float* __restrict__ bias, u16* __restrict__ Cb,
    float* __restrict__ Cf, int M, int N, int K, int act) {
  __shared__ __align__(16) u16 sA[BM * LDT];
  __shared__ __align__(16) u16 sB[BN * LDT];

  const int tid  = threadIdx.x;
  const int lane = tid & 31;
  const int wid  = tid >> 5;
  const int wy   = wid >> 1, wx = wid & 1;   // 4x2 wave grid
  const int m0 = blockIdx.y * BM, n0 = blockIdx.x * BN;

  v8f acc[TM][TN];
#pragma unroll
  for (int i = 0; i < TM; ++i)
#pragma unroll
    for (int j = 0; j < TN; ++j) acc[i][j] = vzero();

  const int lr = tid >> 1;          // 0..127 : tile row to stage
  const int lc = (tid & 1) * 16;    // 0/16   : half-row (16 halves = 32B)
  const int fr = lane & 15;         // fragment row
  const int kc = (lane >> 4) * 8;   // fragment K chunk

  for (int k0 = 0; k0 < K; k0 += BK) {
    // Async-stage 128x32 A tile and 128x32 Bt tile straight into LDS.
    const u16* ga = A + (size_t)(m0 + lr) * K + k0 + lc;
    u16* da = sA + lr * LDT + lc;
    async_ld_b128(da, ga);
    async_ld_b128(da + 8, ga + 8);
    const u16* gb = Bt + (size_t)(n0 + lr) * K + k0 + lc;
    u16* db = sB + lr * LDT + lc;
    async_ld_b128(db, gb);
    async_ld_b128(db + 8, gb + 8);
    if (k0 + BK < K) {  // prefetch next K tile into L2
      __builtin_prefetch(ga + BK, 0, 3);
      __builtin_prefetch(gb + BK, 0, 3);
    }
    wait_async0();
    __syncthreads();

    v16bf af[TM], bf[TN];
#pragma unroll
    for (int tm = 0; tm < TM; ++tm)
      af[tm] = load_frag(sA + (wy * 32 + tm * 16 + fr) * LDT + kc);
#pragma unroll
    for (int tn = 0; tn < TN; ++tn)
      bf[tn] = load_frag(sB + (wx * 64 + tn * 16 + fr) * LDT + kc);
#pragma unroll
    for (int tm = 0; tm < TM; ++tm)
#pragma unroll
      for (int tn = 0; tn < TN; ++tn)
        acc[tm][tn] = wmma_bf16(af[tm], bf[tn], acc[tm][tn]);
    __syncthreads();
  }

  // Epilogue: bias + activation + store.  C layout: VGPR r, lane l ->
  // row = r + 8*(l>>4), col = l&15 within each 16x16 tile.
#pragma unroll
  for (int tm = 0; tm < TM; ++tm)
#pragma unroll
    for (int tn = 0; tn < TN; ++tn) {
      const int rowb = m0 + wy * 32 + tm * 16 + ((lane >> 4) << 3);
      const int col  = n0 + wx * 64 + tn * 16 + (lane & 15);
      const float bc = bias[col];
#pragma unroll
      for (int r = 0; r < 8; ++r) {
        float v = acc[tm][tn][r] + bc;
        if (act == 1) v = 0.5f * v * (1.0f + erff(v * 0.70710678118654752f));
        const size_t idx = (size_t)(rowb + r) * N + col;
        if (Cb) Cb[idx] = f2bf(v);
        if (Cf) Cf[idx] = v;
      }
    }
}

// ---------------------------------------------------------------------------
// Causal flash attention: one wave (32 threads) per 16-query block per (b,h).
// S = Q Kᵀ (4 WMMAs / 32 keys), online softmax in f32, P staged through LDS
// to re-shape C-layout -> A-fragment, then O += P V (4 WMMAs / 32 keys).
// Q,K bf16 [B*L][D]; Vt bf16 [(b*H+h)*Hd][L]; Ob bf16 [B*L][D].
// ---------------------------------------------------------------------------
__global__ __launch_bounds__(32) void k_attn_fa(const u16* __restrict__ Q,
                                                const u16* __restrict__ Kc,
                                                const u16* __restrict__ Vt,
                                                u16* __restrict__ Ob) {
  constexpr int L = 2048, D = 1024, H = 16, Hd = 64;
  __shared__ __align__(16) u16 pbuf[16 * 32];

  const int lane = threadIdx.x;
  const int nqb  = L / 16;
  const int qblk = blockIdx.x % nqb;
  const int h    = (blockIdx.x / nqb) % H;
  const int b    = blockIdx.x / (nqb * H);
  const int q0   = qblk * 16;
  const int fr   = lane & 15;
  const int kc   = (lane >> 4) * 8;

  const u16* qrow = Q + (size_t)(b * L + q0 + fr) * D + h * Hd;
  const v16bf qf0 = load_frag(qrow + kc);         // Hd 0..31
  const v16bf qf1 = load_frag(qrow + 32 + kc);    // Hd 32..63

  float mrow[8], lrow[8];
  v8f o[4];
#pragma unroll
  for (int r = 0; r < 8; ++r) { mrow[r] = -1e30f; lrow[r] = 0.f; }
#pragma unroll
  for (int c = 0; c < 4; ++c) o[c] = vzero();

  const float scale = 0.125f;  // 1/sqrt(Hd)

  for (int k0 = 0; k0 < q0 + 16; k0 += 32) {
    const u16* krA = Kc + (size_t)(b * L + k0 + fr) * D + h * Hd;
    const u16* krB = krA + (size_t)16 * D;
    v8f s0 = vzero(), s1 = vzero();
    s0 = wmma_bf16(qf0, load_frag(krA + kc),      s0);
    s0 = wmma_bf16(qf1, load_frag(krA + 32 + kc), s0);
    s1 = wmma_bf16(qf0, load_frag(krB + kc),      s1);
    s1 = wmma_bf16(qf1, load_frag(krB + 32 + kc), s1);

#pragma unroll
    for (int r = 0; r < 8; ++r) {
      const int Mi = r + ((lane >> 4) << 3);
      const int qg = q0 + Mi;
      float v0 = s0[r] * scale;
      float v1 = s1[r] * scale;
      if (k0 + fr > qg)      v0 = -1e30f;   // causal mask
      if (k0 + 16 + fr > qg) v1 = -1e30f;
      const float mx = rmax16(fmaxf(v0, v1));
      const float mn = fmaxf(mrow[r], mx);
      const float al = __expf(mrow[r] - mn);
      mrow[r] = mn;
      const float p0 = __expf(v0 - mn);
      const float p1 = __expf(v1 - mn);
      lrow[r] = lrow[r] * al + rsum16(p0 + p1);
      o[0][r] *= al; o[1][r] *= al; o[2][r] *= al; o[3][r] *= al;
      pbuf[Mi * 32 + fr]      = f2bf(p0);
      pbuf[Mi * 32 + 16 + fr] = f2bf(p1);
    }
    // DS ops are in-order per wave; wait drains the stores before reload.
    asm volatile("s_wait_dscnt 0x0" ::: "memory");
    const v16bf pf = load_frag(pbuf + fr * 32 + kc);
#pragma unroll
    for (int c = 0; c < 4; ++c) {
      const u16* vrow =
          Vt + (size_t)((b * H + h) * Hd + c * 16 + fr) * L + k0;
      o[c] = wmma_bf16(pf, load_frag(vrow + kc), o[c]);
    }
  }

#pragma unroll
  for (int c = 0; c < 4; ++c)
#pragma unroll
    for (int r = 0; r < 8; ++r) {
      const int Mi = r + ((lane >> 4) << 3);
      const float val = o[c][r] / lrow[r];
      Ob[(size_t)(b * L + q0 + Mi) * D + h * Hd + c * 16 + fr] = f2bf(val);
    }
}

// ---------------------------------------------------------------------------
// Residual + LayerNorm over D=1024: out = LN(a + r) * g + be.
// block 256 (each thread 4 elems), grid = rows.  Optional bf16 copy.
// ---------------------------------------------------------------------------
__global__ __launch_bounds__(256) void k_resid_ln(
    const float* __restrict__ a, const float* __restrict__ r,
    const float* __restrict__ g, const float* __restrict__ be,
    float* __restrict__ of, u16* __restrict__ ob, int D) {
  __shared__ float red[256];
  __shared__ float stats[2];
  const int row = blockIdx.x, t = threadIdx.x;
  const float* pa = a + (size_t)row * D;
  const float* pr = r + (size_t)row * D;

  float v[4];
  float s = 0.f, s2 = 0.f;
#pragma unroll
  for (int j = 0; j < 4; ++j) {
    v[j] = pa[t + j * 256] + pr[t + j * 256];
    s += v[j];
    s2 += v[j] * v[j];
  }
  red[t] = s;
  __syncthreads();
  for (int off = 128; off > 0; off >>= 1) {
    if (t < off) red[t] += red[t + off];
    __syncthreads();
  }
  if (t == 0) stats[0] = red[0];
  __syncthreads();
  red[t] = s2;
  __syncthreads();
  for (int off = 128; off > 0; off >>= 1) {
    if (t < off) red[t] += red[t + off];
    __syncthreads();
  }
  if (t == 0) stats[1] = red[0];
  __syncthreads();

  const float mu  = stats[0] / (float)D;
  const float var = stats[1] / (float)D - mu * mu;
  const float rs  = rsqrtf(var + 1e-5f);
#pragma unroll
  for (int j = 0; j < 4; ++j) {
    const int c = t + j * 256;
    const float y = (v[j] - mu) * rs * g[c] + be[c];
    of[(size_t)row * D + c] = y;
    if (ob) ob[(size_t)row * D + c] = f2bf(y);
  }
}

// ---------------------------------------------------------------------------
// Host orchestration
// ---------------------------------------------------------------------------
extern "C" void kernel_launch(void* const* d_in, const int* in_sizes, int n_in,
                              void* d_out, int out_size, void* d_ws,
                              size_t ws_size, hipStream_t stream) {
  (void)in_sizes; (void)n_in; (void)out_size; (void)ws_size;
  constexpr int Bq = 2, L = 2048, D = 1024, H = 16, Hd = 64, F = 4096;
  constexpr int M = Bq * L;  // 4096 token rows

  const float* x   = (const float*)d_in[0];
  const float* Wq  = (const float*)d_in[1];
  const float* bq  = (const float*)d_in[2];
  const float* Wk  = (const float*)d_in[3];
  const float* bk  = (const float*)d_in[4];
  const float* Wv  = (const float*)d_in[5];
  const float* bv  = (const float*)d_in[6];
  const float* Wo  = (const float*)d_in[7];
  const float* bo  = (const float*)d_in[8];
  const float* W1  = (const float*)d_in[9];
  const float* b1  = (const float*)d_in[10];
  const float* W2  = (const float*)d_in[11];
  const float* b2  = (const float*)d_in[12];
  const float* g1  = (const float*)d_in[13];
  const float* be1 = (const float*)d_in[14];
  const float* g2  = (const float*)d_in[15];
  const float* be2 = (const float*)d_in[16];
  float* out = (float*)d_out;

  char* wp = (char*)d_ws;
  auto alloc = [&](size_t bytes) -> void* {
    void* p = (void*)wp;
    wp += (bytes + 255) & ~(size_t)255;
    return p;
  };
  u16* xb   = (u16*)alloc((size_t)M * D * 2);   // bf16 activations
  u16* wqT  = (u16*)alloc((size_t)D * D * 2);   // transposed bf16 weights
  u16* wkT  = (u16*)alloc((size_t)D * D * 2);
  u16* wvT  = (u16*)alloc((size_t)D * D * 2);
  u16* woT  = (u16*)alloc((size_t)D * D * 2);
  u16* w1T  = (u16*)alloc((size_t)F * D * 2);
  u16* w2T  = (u16*)alloc((size_t)D * F * 2);
  u16* qb   = (u16*)alloc((size_t)M * D * 2);
  u16* kb   = (u16*)alloc((size_t)M * D * 2);
  u16* vb   = (u16*)alloc((size_t)M * D * 2);
  u16* vt   = (u16*)alloc((size_t)Bq * H * Hd * L * 2);
  u16* ab   = (u16*)alloc((size_t)M * D * 2);   // attention output (bf16)
  float* pr = (float*)alloc((size_t)M * D * 4); // proj / ffn-out (f32, reused)
  float* x1 = (float*)alloc((size_t)M * D * 4); // after first LN (f32)
  u16* x1b  = (u16*)alloc((size_t)M * D * 2);
  u16* hb   = (u16*)alloc((size_t)M * F * 2);   // FFN hidden (bf16)

  const dim3 t256(256), t32x8(32, 8), t32(32);

  // 1) activations + weights -> bf16 (weights transposed to [N][K])
  k_f32_to_bf16<<<4096, t256, 0, stream>>>(x, xb, M * D);
  k_wtrans<<<dim3(D / 32, D / 32), t32x8, 0, stream>>>(Wq, wqT, D, D);
  k_wtrans<<<dim3(D / 32, D / 32), t32x8, 0, stream>>>(Wk, wkT, D, D);
  k_wtrans<<<dim3(D / 32, D / 32), t32x8, 0, stream>>>(Wv, wvT, D, D);
  k_wtrans<<<dim3(D / 32, D / 32), t32x8, 0, stream>>>(Wo, woT, D, D);
  k_wtrans<<<dim3(D / 32, F / 32), t32x8, 0, stream>>>(W1, w1T, D, F);
  k_wtrans<<<dim3(F / 32, D / 32), t32x8, 0, stream>>>(W2, w2T, F, D);

  // 2) QKV projections (bf16 out)
  k_gemm_bf16<<<dim3(D / BN, M / BM), t256, 0, stream>>>(
      xb, wqT, bq, qb, nullptr, M, D, D, 0);
  k_gemm_bf16<<<dim3(D / BN, M / BM), t256, 0, stream>>>(
      xb, wkT, bk, kb, nullptr, M, D, D, 0);
  k_gemm_bf16<<<dim3(D / BN, M / BM), t256, 0, stream>>>(
      xb, wvT, bv, vb, nullptr, M, D, D, 0);
  k_vtrans<<<dim3(L / 32, Hd / 32, Bq * H), t32x8, 0, stream>>>(
      vb, vt, H, L, D, Hd);

  // 3) causal flash attention
  k_attn_fa<<<dim3(Bq * H * (L / 16)), t32, 0, stream>>>(qb, kb, vt, ab);

  // 4) output projection (f32 out) + residual LN1
  k_gemm_bf16<<<dim3(D / BN, M / BM), t256, 0, stream>>>(
      ab, woT, bo, nullptr, pr, M, D, D, 0);
  k_resid_ln<<<M, t256, 0, stream>>>(x, pr, g1, be1, x1, x1b, D);

  // 5) FFN: GELU(x1 @ W1 + b1) @ W2 + b2, then residual LN2 -> out
  k_gemm_bf16<<<dim3(F / BN, M / BM), t256, 0, stream>>>(
      x1b, w1T, b1, hb, nullptr, M, F, D, 1);
  k_gemm_bf16<<<dim3(D / BN, M / BM), t256, 0, stream>>>(
      hb, w2T, b2, nullptr, pr, M, D, F, 0);
  k_resid_ln<<<M, t256, 0, stream>>>(x1, pr, g2, be2, out, nullptr, D);
}
